// CausalSelfAttention_57088705298768
// MI455X (gfx1250) — compile-verified, matrix operand order checked by reading, not run on hardware
//
#include <hip/hip_runtime.h>
#include <hip/hip_bf16.h>
#include <stdint.h>
#include <stddef.h>

// ---------------------------------------------------------------------------
// Types
// ---------------------------------------------------------------------------
typedef __bf16 bf16_t;
typedef __attribute__((ext_vector_type(16))) __bf16 v16bf;
typedef __attribute__((ext_vector_type(8)))  float  v8f;
typedef __attribute__((ext_vector_type(4)))  unsigned int v4u;
typedef __attribute__((ext_vector_type(4)))  float  v4f;

union Frag16 {            // 16 bf16 elements = one WMMA A/B fragment (wave32)
    v16bf v;
    v4u   q[2];
};

// ---------------------------------------------------------------------------
// Problem dims (fixed by the reference)
// ---------------------------------------------------------------------------
#define MB    2
#define SEQ   2048
#define DM    1024
#define NH    16
#define HD    64
#define MTOT  (MB * SEQ)      // 4096 rows

// GEMM tiling
#define BM    128
#define BN    128
#define BK    32
#define AST   40              // LDS row stride in bf16 elems (80B, 16B-aligned)

// Attention tiling
#define QT    128             // q rows per block (8 waves x 16)
#define KVT   32              // kv rows per tile
#define KSTR  72              // K tile LDS stride (144B, 16B-aligned)
#define VSTR  40              // V^T tile LDS stride
#define PSTR  40              // per-wave P tile LDS stride

// LDS byte address for gfx1250 LDS-targeting DMA: generic shared pointers
// carry the wave-relative LDS offset in addr[31:0] (flat aperture decode).
__device__ __forceinline__ unsigned lds_addr32(const void* p)
{
    return (unsigned)(uintptr_t)p;
}

// ---------------------------------------------------------------------------
// Tensor Data Mover (CDNA5): one instruction moves a padded 2D bf16 tile
// global -> LDS, tracked by TENSORcnt. Issued by one wave; EXEC is ignored.
// ---------------------------------------------------------------------------
#if __has_builtin(__builtin_amdgcn_tensor_load_to_lds)
#define HAVE_TDM 1
#else
#define HAVE_TDM 0
#endif

#if HAVE_TDM
typedef __attribute__((ext_vector_type(4))) unsigned int u32x4;
typedef __attribute__((ext_vector_type(8))) int          i32x8;
typedef __attribute__((ext_vector_type(4))) int          i32x4;

// tileW/tileH in elements; tensorStride = row pitch in elements.
// padIntervalCode: pad after 2<<code DWORDs; padAmountCode: pad (code+1) DWORDs.
__device__ __forceinline__ void tdm_load_tile_bf16(
    unsigned ldsAddr, const void* gptr,
    unsigned tileW, unsigned tileH, unsigned tensorStride,
    unsigned padIntervalCode, unsigned padAmountCode)
{
    const unsigned long long ga = (unsigned long long)(uintptr_t)gptr;

    u32x4 g0;
    g0[0] = 1u;                                               // count=1, user D#
    g0[1] = __builtin_amdgcn_readfirstlane(ldsAddr);          // lds_addr
    g0[2] = __builtin_amdgcn_readfirstlane((unsigned)ga);     // global_addr lo
    g0[3] = __builtin_amdgcn_readfirstlane(
                ((unsigned)(ga >> 32) & 0x01FFFFFFu) | (2u << 30)); // hi | type=2

    i32x8 g1;
    g1[0] = (int)((1u << 16) |                 // data_size = 2B
                  (1u << 20) |                 // pad_enable
                  (padIntervalCode << 22) | (padAmountCode << 25));
    g1[1] = (int)((tensorStride & 0xFFFFu) << 16);            // tensor_dim0 lo16
    g1[2] = (int)((tensorStride >> 16) | (4096u << 16));      // dim0 hi | dim1 lo
    g1[3] = (int)(tileW << 16);                               // dim1 hi=0 | tile_dim0
    g1[4] = (int)tileH;                                       // tile_dim1, tile_dim2=0
    g1[5] = (int)tensorStride;                                // dim0_stride lo32
    g1[6] = 0;                                                // stride hi / dim1_stride
    g1[7] = 0;

    const i32x4 z4 = {0, 0, 0, 0};
#if defined(__clang_major__) && (__clang_major__ >= 23)
    const i32x8 z8 = {0, 0, 0, 0, 0, 0, 0, 0};
    __builtin_amdgcn_tensor_load_to_lds(g0, g1, z4, z4, z8, 0);
#else
    __builtin_amdgcn_tensor_load_to_lds(g0, g1, z4, z4, 0);
#endif
}
#endif  // HAVE_TDM

// ---------------------------------------------------------------------------
// fp32 -> bf16 conversion (vectorized; n is always a multiple of 4 here)
// ---------------------------------------------------------------------------
__global__ __launch_bounds__(256, 1)
void cvt_f32_to_bf16(const float* __restrict__ s, bf16_t* __restrict__ d, int n)
{
    int i = (blockIdx.x * 256 + threadIdx.x) * 4;
    if (i >= n) return;
    v4f v = *(const v4f*)(s + i);
    d[i + 0] = (bf16_t)v.x;
    d[i + 1] = (bf16_t)v.y;
    d[i + 2] = (bf16_t)v.z;
    d[i + 3] = (bf16_t)v.w;
}

// ---------------------------------------------------------------------------
// C[M][N] = scale * (A[M][K] @ W[N][K]^T), bf16 inputs, f32 accum.
// Tile fills: TDM tensor_load_to_lds (wave 0, double buffered); fallback is
// per-thread GLOBAL_LOAD_ASYNC_TO_LDS_B128.
// ---------------------------------------------------------------------------
__global__ __launch_bounds__(256, 1)
void gemm_bf16_wmma(const bf16_t* __restrict__ A,
                    const bf16_t* __restrict__ W,
                    void* __restrict__ C,
                    int M, int N, int K, float scale, int f32out)
{
    __shared__ bf16_t sA[2][BM * AST];
    __shared__ bf16_t sB[2][BN * AST];

    const int t     = threadIdx.x;
    const int mBase = blockIdx.x * BM;
    const int nBase = blockIdx.y * BN;

    // wave layout: 8 waves -> 2 (M, 64 rows) x 4 (N, 32 cols)
    const int wid  = t >> 5;
    const int lane = t & 31;
    const int hi   = lane >> 4;           // half-wave select (K-offset 8)
    const int r    = lane & 15;           // fragment row index
    const int m0   = (wid >> 2) * 64;
    const int n0   = (wid & 3) * 32;

    const v8f zero = {0.f, 0.f, 0.f, 0.f, 0.f, 0.f, 0.f, 0.f};
    v8f acc[4][2];
#pragma unroll
    for (int i = 0; i < 4; ++i)
#pragma unroll
        for (int j = 0; j < 2; ++j) acc[i][j] = zero;

    const int KT = K / BK;

#if HAVE_TDM
    // one TDM descriptor per matrix per tile; rows of 32 bf16 (16 DW) padded
    // by 4 DW -> LDS stride 40 elems == AST
    auto issue_tile = [&](int buf, int k0) {
        tdm_load_tile_bf16(lds_addr32(&sA[buf][0]),
                           A + (size_t)mBase * K + k0, BK, BM, K, 3, 3);
        tdm_load_tile_bf16(lds_addr32(&sB[buf][0]),
                           W + (size_t)nBase * K + k0, BK, BN, K, 3, 3);
    };
    if (wid == 0) issue_tile(0, 0);
#else
    const int lrow  = t >> 1;             // 0..127
    const int lhalf = (t & 1) * 16;       // 0 or 16 elems
    auto issue_tile = [&](int buf, int k0) {
        const bf16_t* gA = A + (size_t)(mBase + lrow) * K + k0 + lhalf;
        const bf16_t* gB = W + (size_t)(nBase + lrow) * K + k0 + lhalf;
        unsigned la = lds_addr32(&sA[buf][lrow * AST + lhalf]);
        unsigned lb = lds_addr32(&sB[buf][lrow * AST + lhalf]);
        asm volatile(
            "global_load_async_to_lds_b128 %0, %2, off\n\t"
            "global_load_async_to_lds_b128 %0, %2, off offset:16\n\t"
            "global_load_async_to_lds_b128 %1, %3, off\n\t"
            "global_load_async_to_lds_b128 %1, %3, off offset:16"
            :: "v"(la), "v"(lb),
               "v"((unsigned long long)(uintptr_t)gA),
               "v"((unsigned long long)(uintptr_t)gB)
            : "memory");
    };
    issue_tile(0, 0);
#endif

    for (int kt = 0; kt < KT; ++kt) {
        const int  cur = kt & 1;
        const bool pre = (kt + 1) < KT;
#if HAVE_TDM
        if (wid == 0) {
            if (pre) {
                // cur^1 was fully consumed before the trailing barrier of the
                // previous iteration -> safe to refill now
                issue_tile(cur ^ 1, (kt + 1) * BK);
                // TDM ops of one wave complete in order: <=2 outstanding
                // means tile kt has landed
                __builtin_amdgcn_s_wait_tensorcnt(2);
            } else {
                __builtin_amdgcn_s_wait_tensorcnt(0);
            }
        }
#else
        if (pre) {
            issue_tile(cur ^ 1, (kt + 1) * BK);
            asm volatile("s_wait_asynccnt 4" ::: "memory");
        } else {
            asm volatile("s_wait_asynccnt 0" ::: "memory");
        }
#endif
        __syncthreads();   // tile kt visible to every wave

        // fragment loads per ISA 16-bit striping: chunks at hi*8 and 16+hi*8
        Frag16 af[4], bfr[2];
#pragma unroll
        for (int i = 0; i < 4; ++i) {
            const bf16_t* p = &sA[cur][(m0 + i * 16 + r) * AST + hi * 8];
            af[i].q[0] = *(const v4u*)p;
            af[i].q[1] = *(const v4u*)(p + 16);
        }
#pragma unroll
        for (int j = 0; j < 2; ++j) {
            const bf16_t* p = &sB[cur][(n0 + j * 16 + r) * AST + hi * 8];
            bfr[j].q[0] = *(const v4u*)p;
            bfr[j].q[1] = *(const v4u*)(p + 16);
        }
#pragma unroll
        for (int i = 0; i < 4; ++i)
#pragma unroll
            for (int j = 0; j < 2; ++j)
                acc[i][j] = __builtin_amdgcn_wmma_f32_16x16x32_bf16(
                    false, af[i].v, false, bfr[j].v, (short)0, acc[i][j],
                    false, false);

        __syncthreads();   // all waves done reading cur before it is refilled
    }

    // epilogue: C-matrix layout lane(0..15)=col, vgpr rr=row (+8 for hi half)
#pragma unroll
    for (int i = 0; i < 4; ++i)
#pragma unroll
        for (int j = 0; j < 2; ++j)
#pragma unroll
            for (int rr = 0; rr < 8; ++rr) {
                const int row = mBase + m0 + i * 16 + hi * 8 + rr;
                const int col = nBase + n0 + j * 16 + r;
                const float val = acc[i][j][rr] * scale;
                if (f32out) ((float*)C)[(size_t)row * N + col] = val;
                else        ((bf16_t*)C)[(size_t)row * N + col] = (bf16_t)val;
            }
}

// ---------------------------------------------------------------------------
// Flash attention, causal. Q already scaled by 1/sqrt(HD).
// Q/K/V layout: [b*SEQ + s][h*HD + d] bf16. Output same layout (merged heads).
// Block: 128 q rows for one (b,h); 8 waves each own a 16-row strip.
// K tile filled by TDM (async fallback); V transposed through VGPRs.
// ---------------------------------------------------------------------------
__global__ __launch_bounds__(256, 1)
void attn_wmma(const bf16_t* __restrict__ Q,
               const bf16_t* __restrict__ Kc,
               const bf16_t* __restrict__ V,
               bf16_t* __restrict__ O)
{
    __shared__ bf16_t sK[KVT * KSTR];       // K tile   [kv][d]
    __shared__ bf16_t sVt[HD * VSTR];       // V^T tile [d][kv]
    __shared__ bf16_t sP[8][16 * PSTR];     // per-wave P re-striping buffer

    const int t    = threadIdx.x;
    const int wid  = t >> 5;
    const int lane = t & 31;
    const int hi   = lane >> 4;
    const int r    = lane & 15;

    const int qtile = blockIdx.x;            // 0..SEQ/QT-1
    const int bh    = blockIdx.y;            // 0..MB*NH-1
    const int b     = bh >> 4;
    const int h     = bh & 15;
    const int qbase = qtile * QT;
    const int qrow  = qbase + wid * 16;      // this wave's strip

    // Q fragments for the whole kv loop: 16 rows x 64 d = 2 A-fragments
    Frag16 aq[2];
#pragma unroll
    for (int f = 0; f < 2; ++f) {
        const bf16_t* p = Q + (size_t)(b * SEQ + qrow + r) * DM
                            + h * HD + f * 32 + hi * 8;
        aq[f].q[0] = *(const v4u*)p;
        aq[f].q[1] = *(const v4u*)(p + 16);
    }

    const v8f zero = {0.f, 0.f, 0.f, 0.f, 0.f, 0.f, 0.f, 0.f};
    v8f od[4] = {zero, zero, zero, zero};    // 16 rows x 64 d accumulator
    float mrun[8], lrun[8];
#pragma unroll
    for (int rr = 0; rr < 8; ++rr) { mrun[rr] = -1e30f; lrun[rr] = 0.f; }

    const int ntiles = qbase / KVT + QT / KVT;   // causal upper bound

    for (int kv = 0; kv < ntiles; ++kv) {
        const int kvbase = kv * KVT;

        // K tile straight to LDS. Rows of 64 bf16 (32 DW) padded by 4 DW ->
        // stride 72 elems == KSTR.
#if HAVE_TDM
        if (wid == 0)
            tdm_load_tile_bf16(lds_addr32(&sK[0]),
                               Kc + (size_t)(b * SEQ + kvbase) * DM + h * HD,
                               HD, KVT, DM, 4, 3);
#endif
        {
            const int krow  = t >> 3;         // 0..31
            const int chunk = (t & 7) * 8;    // d offset, 8 elems = 16B
#if !HAVE_TDM
            const bf16_t* gK = Kc + (size_t)(b * SEQ + kvbase + krow) * DM
                                  + h * HD + chunk;
            unsigned lk = lds_addr32(&sK[krow * KSTR + chunk]);
            asm volatile("global_load_async_to_lds_b128 %0, %1, off"
                         :: "v"(lk),
                            "v"((unsigned long long)(uintptr_t)gK)
                         : "memory");
#endif
            // V transposed into LDS so PV B-fragments are row-contiguous
            union { v4u u; bf16_t e[8]; } vd;
            vd.u = *(const v4u*)(V + (size_t)(b * SEQ + kvbase + krow) * DM
                                   + h * HD + chunk);
#pragma unroll
            for (int e2 = 0; e2 < 8; ++e2)
                sVt[(chunk + e2) * VSTR + krow] = vd.e[e2];
        }
#if HAVE_TDM
        if (wid == 0) __builtin_amdgcn_s_wait_tensorcnt(0);
#else
        asm volatile("s_wait_asynccnt 0" ::: "memory");
#endif
        __syncthreads();

        // S = Q K^T  (16 x 32 tile = two f32 C-fragments, K-dim = d = 64)
        v8f s0 = zero, s1 = zero;
#pragma unroll
        for (int f = 0; f < 2; ++f) {
            Frag16 bk0, bk1;
            const bf16_t* p0 = &sK[(0 * 16 + r) * KSTR + f * 32 + hi * 8];
            bk0.q[0] = *(const v4u*)p0;  bk0.q[1] = *(const v4u*)(p0 + 16);
            const bf16_t* p1 = &sK[(1 * 16 + r) * KSTR + f * 32 + hi * 8];
            bk1.q[0] = *(const v4u*)p1;  bk1.q[1] = *(const v4u*)(p1 + 16);
            s0 = __builtin_amdgcn_wmma_f32_16x16x32_bf16(
                false, aq[f].v, false, bk0.v, (short)0, s0, false, false);
            s1 = __builtin_amdgcn_wmma_f32_16x16x32_bf16(
                false, aq[f].v, false, bk1.v, (short)0, s1, false, false);
        }

        // causal mask + online softmax. Row (rr + hi*8) lives in a 16-lane
        // half, so xor-shuffles with masks 1,2,4,8 reduce exactly that row.
        float alpha[8];
#pragma unroll
        for (int rr = 0; rr < 8; ++rr) {
            const int qi = qrow + hi * 8 + rr;
            const int k0 = kvbase + r;
            const int k1 = kvbase + 16 + r;
            float v0 = (k0 <= qi) ? s0[rr] : -1e30f;
            float v1 = (k1 <= qi) ? s1[rr] : -1e30f;
            float tm = fmaxf(v0, v1);
            tm = fmaxf(tm, __shfl_xor(tm, 1, 32));
            tm = fmaxf(tm, __shfl_xor(tm, 2, 32));
            tm = fmaxf(tm, __shfl_xor(tm, 4, 32));
            tm = fmaxf(tm, __shfl_xor(tm, 8, 32));
            const float mnew = fmaxf(mrun[rr], tm);
            alpha[rr] = __expf(mrun[rr] - mnew);
            const float p0e = __expf(v0 - mnew);
            const float p1e = __expf(v1 - mnew);
            s0[rr] = p0e;
            s1[rr] = p1e;
            float ls = p0e + p1e;
            ls += __shfl_xor(ls, 1, 32);
            ls += __shfl_xor(ls, 2, 32);
            ls += __shfl_xor(ls, 4, 32);
            ls += __shfl_xor(ls, 8, 32);
            lrun[rr] = lrun[rr] * alpha[rr] + ls;
            mrun[rr] = mnew;
        }
#pragma unroll
        for (int f = 0; f < 4; ++f)
#pragma unroll
            for (int rr = 0; rr < 8; ++rr) od[f][rr] *= alpha[rr];

        // re-stripe P from C-layout to A-layout via per-wave LDS (in-order DS)
#pragma unroll
        for (int rr = 0; rr < 8; ++rr) {
            sP[wid][(hi * 8 + rr) * PSTR + r]      = (bf16_t)s0[rr];
            sP[wid][(hi * 8 + rr) * PSTR + 16 + r] = (bf16_t)s1[rr];
        }
        Frag16 pf;
        {
            const bf16_t* p = &sP[wid][r * PSTR + hi * 8];
            pf.q[0] = *(const v4u*)p;
            pf.q[1] = *(const v4u*)(p + 16);
        }

        // O += P V   (K-dim = 32 kv; 4 fragments across d)
#pragma unroll
        for (int f = 0; f < 4; ++f) {
            Frag16 vf;
            const bf16_t* p = &sVt[(f * 16 + r) * VSTR + hi * 8];
            vf.q[0] = *(const v4u*)p;
            vf.q[1] = *(const v4u*)(p + 16);
            od[f] = __builtin_amdgcn_wmma_f32_16x16x32_bf16(
                false, pf.v, false, vf.v, (short)0, od[f], false, false);
        }
        __syncthreads();   // all reads of sK/sVt done before next tile fill
    }

    // normalize and store merged-head bf16 output
#pragma unroll
    for (int f = 0; f < 4; ++f)
#pragma unroll
        for (int rr = 0; rr < 8; ++rr) {
            const int qi = qrow + hi * 8 + rr;
            const float val = od[f][rr] / lrun[rr];
            O[(size_t)(b * SEQ + qi) * DM + h * HD + f * 16 + r] = (bf16_t)val;
        }
}

// ---------------------------------------------------------------------------
// Launch: cvt -> QKV GEMMs -> flash attention -> output GEMM
// Workspace layout (bytes):
//   [0,8M)   Xb      bf16 x
//   [8M..)   Wqb/Wkb/Wvb/Wob  (2 MiB each)
//   [16M..)  Qb, Kb, Vb (8 MiB each)
//   [40M..)  Ab (merged attention output, 8 MiB)      total 48 MiB
// ---------------------------------------------------------------------------
extern "C" void kernel_launch(void* const* d_in, const int* in_sizes, int n_in,
                              void* d_out, int out_size, void* d_ws, size_t ws_size,
                              hipStream_t stream)
{
    (void)in_sizes; (void)n_in; (void)out_size; (void)ws_size;
    const float* x  = (const float*)d_in[0];
    const float* Wq = (const float*)d_in[1];
    const float* Wk = (const float*)d_in[2];
    const float* Wv = (const float*)d_in[3];
    const float* Wo = (const float*)d_in[4];

    char* ws = (char*)d_ws;
    bf16_t* Xb  = (bf16_t*)(ws);
    bf16_t* Wqb = (bf16_t*)(ws + ( 8u << 20));
    bf16_t* Wkb = (bf16_t*)(ws + (10u << 20));
    bf16_t* Wvb = (bf16_t*)(ws + (12u << 20));
    bf16_t* Wob = (bf16_t*)(ws + (14u << 20));
    bf16_t* Qb  = (bf16_t*)(ws + (16u << 20));
    bf16_t* Kb  = (bf16_t*)(ws + (24u << 20));
    bf16_t* Vb  = (bf16_t*)(ws + (32u << 20));
    bf16_t* Ab  = (bf16_t*)(ws + (40u << 20));

    const int nx = MTOT * DM;   // 4,194,304
    const int nw = DM * DM;     // 1,048,576
    cvt_f32_to_bf16<<<nx / 1024, 256, 0, stream>>>(x,  Xb,  nx);
    cvt_f32_to_bf16<<<nw / 1024, 256, 0, stream>>>(Wq, Wqb, nw);
    cvt_f32_to_bf16<<<nw / 1024, 256, 0, stream>>>(Wk, Wkb, nw);
    cvt_f32_to_bf16<<<nw / 1024, 256, 0, stream>>>(Wv, Wvb, nw);
    cvt_f32_to_bf16<<<nw / 1024, 256, 0, stream>>>(Wo, Wob, nw);

    dim3 g(MTOT / BM, DM / BN);   // 32 x 8
    // softmax 1/sqrt(HD) folded into Q
    gemm_bf16_wmma<<<g, 256, 0, stream>>>(Xb, Wqb, Qb, MTOT, DM, DM, 0.125f, 0);
    gemm_bf16_wmma<<<g, 256, 0, stream>>>(Xb, Wkb, Kb, MTOT, DM, DM, 1.0f,   0);
    gemm_bf16_wmma<<<g, 256, 0, stream>>>(Xb, Wvb, Vb, MTOT, DM, DM, 1.0f,   0);

    attn_wmma<<<dim3(SEQ / QT, MB * NH), 256, 0, stream>>>(Qb, Kb, Vb, Ab);

    gemm_bf16_wmma<<<g, 256, 0, stream>>>(Ab, Wob, d_out, MTOT, DM, DM, 1.0f, 1);
}
